// CrossEncoder_6914897347349
// MI455X (gfx1250) — compile-verified, matrix operand order checked by reading, not run on hardware
//
#include <hip/hip_runtime.h>

#define B_  4
#define QT_ 1024
#define N_  2048
#define E_  512
#define H_  8
#define DH_ 64

typedef __attribute__((ext_vector_type(16))) __bf16 v16bf;
typedef __attribute__((ext_vector_type(16))) short  v16s;
typedef __attribute__((ext_vector_type(8)))  short  v8s;
typedef __attribute__((ext_vector_type(4)))  short  v4s;
typedef __attribute__((ext_vector_type(8)))  float  v8f;

// float -> bf16 bits (round-to-nearest-even)
__device__ __forceinline__ unsigned short f2bf(float f) {
    unsigned u = __builtin_bit_cast(unsigned, f);
    unsigned r = u + 0x7FFFu + ((u >> 16) & 1u);
    return (unsigned short)(r >> 16);
}

// CDNA5 16-bit A/B fragment from a K-contiguous 32-element run:
// lane elements 0-7  = p[hi .. hi+7]      (one b128)
// lane elements 8-15 = p[16+hi .. 23+hi]  (one b128)
// hi = 0 for lanes 0-15, 8 for lanes 16-31.
__device__ __forceinline__ v16s frag16(const unsigned short* p, int hi) {
    v8s lo = *(const v8s*)(p + hi);
    v8s hh = *(const v8s*)(p + 16 + hi);
    return __builtin_shufflevector(lo, hh, 0, 1, 2, 3, 4, 5, 6, 7,
                                           8, 9, 10, 11, 12, 13, 14, 15);
}

__device__ __forceinline__ v8f wmma_bf16(v16s a, v16s b, v8f c) {
    return __builtin_amdgcn_wmma_f32_16x16x32_bf16(
        false, __builtin_bit_cast(v16bf, a),
        false, __builtin_bit_cast(v16bf, b),
        (short)0, c, false, false);
}

// ---------- Kernel 0: W (512x512 fp32, row=k) -> Wt (bf16, row=n, K-contiguous) ----------
__global__ __launch_bounds__(256)
void wprep_kernel(const float* __restrict__ W, unsigned short* __restrict__ Wt)
{
    __shared__ unsigned short t[32][33];
    const int tid = threadIdx.x;
    const int n0 = blockIdx.x * 32, k0 = blockIdx.y * 32;
    #pragma unroll
    for (int e = tid; e < 1024; e += 256) {
        int r = e >> 5, c = e & 31;  // r = k, c = n
        t[r][c] = f2bf(W[(size_t)(k0 + r) * E_ + n0 + c]);
    }
    __syncthreads();
    #pragma unroll
    for (int e = tid; e < 1024; e += 256) {
        int r = e >> 5, c = e & 31;  // r = n, c = k
        Wt[(size_t)(n0 + r) * E_ + k0 + c] = t[c][r];
    }
}

// ---------- Kernel 1: X@W^T + b, fused RoPE, heads out bf16 ----------
// A row-block (16x512 bf16, 16 KB) staged ONCE; K-loop barrier-free, fully unrolled,
// software-pipelined (next fragments issued before current WMMA).
// out layout: vt==0 -> (B,H,T,Dh); vt==1 -> transposed (B,H,Dh,T) for K-contiguous V frags.
__global__ __launch_bounds__(256)
void qkv_rope_kernel(const float* __restrict__ X, const unsigned short* __restrict__ Wt,
                     const float* __restrict__ bias, const int* __restrict__ pos,
                     unsigned short* __restrict__ out, int T, int do_rope, int vt)
{
    __shared__ alignas(16) unsigned short aLds[16 * E_];   // 16 KB
    __shared__ float oLds[16 * 128];                       // 8 KB
    const int tid  = threadIdx.x;
    const int wave = tid >> 5, lane = tid & 31, ln = lane & 15;
    const int hi   = (lane >= 16) ? 8 : 0;
    const int m0   = blockIdx.y * 16;
    const int nblk = blockIdx.x * 128;
    const int n0   = nblk + wave * 16;
    const unsigned short* wp = Wt + (size_t)(n0 + ln) * E_;
    const unsigned short* ap = aLds + ln * E_;

    // stage entire 16x512 A block as bf16 (K-contiguous rows)
    #pragma unroll
    for (int e = tid; e < 2048; e += 256) {                // 16 rows x 128 float4 segs
        int r = e >> 7, seg = e & 127;
        const float4 f = *(const float4*)&X[(size_t)(m0 + r) * E_ + seg * 4];
        v4s o;
        o[0] = (short)f2bf(f.x); o[1] = (short)f2bf(f.y);
        o[2] = (short)f2bf(f.z); o[3] = (short)f2bf(f.w);
        *(v4s*)&aLds[r * E_ + seg * 4] = o;
    }
    __syncthreads();

    v8f acc = {};
    v16s av = frag16(ap, hi);
    v16s bv = frag16(wp, hi);
    #pragma unroll
    for (int kk = 0; kk < 16; ++kk) {
        v16s an = av, bn2 = bv;
        if (kk < 15) {                                  // issue next frags before this WMMA
            an  = frag16(ap + kk * 32 + 32, hi);
            bn2 = frag16(wp + kk * 32 + 32, hi);
        }
        acc = wmma_bf16(av, bv, acc);
        av = an; bv = bn2;
    }

    const float bn  = bias[n0 + ln];
    const int   mhi = (lane >= 16) ? 8 : 0;
    #pragma unroll
    for (int r = 0; r < 8; ++r)
        oLds[(r + mhi) * 128 + wave * 16 + ln] = acc[r] + bn;
    __syncthreads();

    for (int e = tid; e < 16 * 128; e += 256) {
        int row = e >> 7, col = e & 127;
        int gcol = nblk + col;
        int h = gcol >> 6, d = gcol & 63;
        int m = m0 + row, b = m / T, t = m % T;
        float val;
        if (do_rope) {
            float x1 = oLds[row * 128 + (col & ~1)];
            float xi = oLds[row * 128 + (col |  1)];
            int   fi = d >> 2;   // reference repeats cos/sin then truncates: pair d uses freq d>>2
            float p  = (float)pos[b * T + t];
            float ang = p * __expf(-0.28782313662425574f * (float)fi);  // base^(-fi/32)
            float c = __cosf(ang), s = __sinf(ang);
            val = (d & 1) ? (x1 * s + xi * c) : (x1 * c - xi * s);
        } else {
            val = oLds[row * 128 + col];
        }
        size_t idx = vt ? ((size_t)(b * H_ + h) * DH_ + d) * T + t
                        : ((size_t)(b * H_ + h) * T + t) * DH_ + d;
        out[idx] = f2bf(val);
    }
}

// ---------- Kernel 2: scores -> softmax -> attn out -> ctx, all rows resident in LDS ----------
__global__ __launch_bounds__(256)
void attn_kernel(const unsigned short* __restrict__ qh, const unsigned short* __restrict__ kh,
                 const unsigned short* __restrict__ vT, const unsigned char* __restrict__ kvmask,
                 float* __restrict__ attn_out, unsigned short* __restrict__ ctx)
{
    extern __shared__ char smem[];
    float*          sc     = (float*)smem;                       // 16 x 2048 fp32 (128 KB)
    unsigned short* scbf   = (unsigned short*)(sc + 16 * N_);    // 16 x 2048 bf16 (64 KB)
    unsigned short* qt     = scbf + 16 * N_;                     // 16 x 64 bf16
    float*          red    = (float*)(qt + 16 * DH_);            // 256 fp32
    float*          rowmax = red + 256;
    float*          rowsum = rowmax + 16;
    float*          part   = sc;                                 // reuse: sc is dead in ctx phase

    const int tid  = threadIdx.x;
    const int wave = tid >> 5, lane = tid & 31, ln = lane & 15;
    const int hi   = (lane >= 16) ? 8 : 0;
    const int mhi  = (lane >= 16) ? 8 : 0;
    const int b = blockIdx.z, h = blockIdx.y, q0 = blockIdx.x * 16;
    const size_t bh = (size_t)(b * H_ + h);

    #pragma unroll
    for (int e = tid; e < 128; e += 256)   // 16 rows x 8-short segs
        *(v8s*)&qt[e * 8] = *(const v8s*)&qh[(bh * QT_ + q0 + (e >> 3)) * DH_ + (e & 7) * 8];
    __syncthreads();

    // --- scores = scale * Q K^T, masked, into LDS (fp32) ---
    const float scale = 0.125f;  // 1/sqrt(Dh)
    {
        // Q fragments are invariant across kv tiles: load once
        const v16s qa0 = frag16(qt + ln * DH_,      hi);
        const v16s qa1 = frag16(qt + ln * DH_ + 32, hi);
        const unsigned short* kp0 = kh + (bh * N_ + ln) * DH_;   // + j*16*DH_ per tile
        v16s b0 = frag16(kp0 + (size_t)wave * 16 * DH_,      hi);
        v16s b1 = frag16(kp0 + (size_t)wave * 16 * DH_ + 32, hi);
        #pragma unroll 2
        for (int i = 0; i < 16; ++i) {
            const int j = wave + i * 8;
            v16s nb0 = b0, nb1 = b1;
            if (i < 15) {                              // next K-tile fragments in flight
                const unsigned short* kpn = kp0 + (size_t)(j + 8) * 16 * DH_;
                nb0 = frag16(kpn,      hi);
                nb1 = frag16(kpn + 32, hi);
            }
            v8f acc = {};
            acc = wmma_bf16(qa0, b0, acc);
            acc = wmma_bf16(qa1, b1, acc);
            int  col = j * 16 + ln;
            bool km  = kvmask[b * N_ + col] != 0;
            #pragma unroll
            for (int r = 0; r < 8; ++r) {
                float s = acc[r] * scale;
                if (!km) s = -1.0e9f;
                sc[(r + mhi) * N_ + col] = s;
            }
            b0 = nb0; b1 = nb1;
        }
    }
    __syncthreads();

    // --- masked softmax over full 2048-wide rows; write attn (fp32) + bf16 LDS copy ---
    {
        const int r = tid >> 4, c = tid & 15;
        float mx = -3.0e38f;
        for (int col = c; col < N_; col += 16) mx = fmaxf(mx, sc[r * N_ + col]);
        red[tid] = mx;
        __syncthreads();
        if (c == 0) {
            float m = red[r << 4];
            #pragma unroll
            for (int i = 1; i < 16; ++i) m = fmaxf(m, red[(r << 4) + i]);
            rowmax[r] = m;
        }
        __syncthreads();
        const float m = rowmax[r];
        float sum = 0.f;
        for (int col = c; col < N_; col += 16) {
            float p = __expf(sc[r * N_ + col] - m);
            if (kvmask[b * N_ + col] == 0) p = 0.f;
            sc[r * N_ + col] = p;
            sum += p;
        }
        red[tid] = sum;
        __syncthreads();
        if (c == 0) {
            float s = 0.f;
            #pragma unroll
            for (int i = 0; i < 16; ++i) s += red[(r << 4) + i];
            rowsum[r] = s;
        }
        __syncthreads();
        const float inv = 1.f / (rowsum[r] + 1e-9f);
        for (int col = c; col < N_; col += 16) {
            float a = sc[r * N_ + col] * inv;
            scbf[r * N_ + col] = f2bf(a);
            attn_out[(bh * QT_ + q0 + r) * N_ + col] = a;   // single HBM write of attn
        }
    }
    __syncthreads();

    // --- ctx = attn @ V, K-dim split across the 8 waves, reduced in LDS ---
    v8f cacc[4] = {{}, {}, {}, {}};
    {
        const unsigned short* vp = vT + (bh * DH_ + ln) * N_;   // + nt*16*N_ per col tile
        int kb = wave * 32;
        v16s av = frag16(scbf + ln * N_ + kb, hi);
        v16s bv0 = frag16(vp + 0 * 16 * N_ + kb, hi);
        v16s bv1 = frag16(vp + 1 * 16 * N_ + kb, hi);
        v16s bv2 = frag16(vp + 2 * 16 * N_ + kb, hi);
        v16s bv3 = frag16(vp + 3 * 16 * N_ + kb, hi);
        #pragma unroll 2
        for (int i = 0; i < 8; ++i) {
            v16s na = av, n0 = bv0, n1 = bv1, n2 = bv2, n3 = bv3;
            if (i < 7) {                               // next K-slice fragments in flight
                const int kn = kb + 256;
                na = frag16(scbf + ln * N_ + kn, hi);
                n0 = frag16(vp + 0 * 16 * N_ + kn, hi);
                n1 = frag16(vp + 1 * 16 * N_ + kn, hi);
                n2 = frag16(vp + 2 * 16 * N_ + kn, hi);
                n3 = frag16(vp + 3 * 16 * N_ + kn, hi);
            }
            cacc[0] = wmma_bf16(av, bv0, cacc[0]);
            cacc[1] = wmma_bf16(av, bv1, cacc[1]);
            cacc[2] = wmma_bf16(av, bv2, cacc[2]);
            cacc[3] = wmma_bf16(av, bv3, cacc[3]);
            av = na; bv0 = n0; bv1 = n1; bv2 = n2; bv3 = n3;
            kb += 256;
        }
    }
    #pragma unroll
    for (int nt = 0; nt < 4; ++nt)
        #pragma unroll
        for (int r = 0; r < 8; ++r)
            part[wave * (16 * DH_) + (r + mhi) * DH_ + nt * 16 + ln] = cacc[nt][r];
    __syncthreads();
    for (int e = tid; e < 16 * DH_; e += 256) {
        float s = 0.f;
        #pragma unroll
        for (int w = 0; w < 8; ++w) s += part[w * (16 * DH_) + e];
        int row = e >> 6, d = e & 63;
        ctx[((size_t)(b * QT_ + q0 + row)) * E_ + h * DH_ + d] = f2bf(s);
    }
}

// ---------- Kernel 3: ctx @ Wo^T + bo, masked by q_mask, fp32 out ----------
__global__ __launch_bounds__(256)
void outproj_kernel(const unsigned short* __restrict__ X, const unsigned short* __restrict__ Wt,
                    const float* __restrict__ bias, const unsigned char* __restrict__ qmask,
                    float* __restrict__ out)
{
    __shared__ alignas(16) unsigned short aLds[16 * E_];   // 16 KB, staged once
    const int tid  = threadIdx.x;
    const int wave = tid >> 5, lane = tid & 31, ln = lane & 15;
    const int hi   = (lane >= 16) ? 8 : 0;
    const int m0 = blockIdx.y * 16;
    const int n0 = blockIdx.x * 128 + wave * 16;
    const unsigned short* wp = Wt + (size_t)(n0 + ln) * E_;
    const unsigned short* ap = aLds + ln * E_;

    #pragma unroll
    for (int e = tid; e < 1024; e += 256) {                // 16 rows x 64 b128 segs
        int r = e >> 6, seg = e & 63;
        *(v8s*)&aLds[r * E_ + seg * 8] = *(const v8s*)&X[(size_t)(m0 + r) * E_ + seg * 8];
    }
    __syncthreads();

    v8f acc = {};
    v16s av = frag16(ap, hi);
    v16s bv = frag16(wp, hi);
    #pragma unroll
    for (int kk = 0; kk < 16; ++kk) {
        v16s an = av, bn2 = bv;
        if (kk < 15) {
            an  = frag16(ap + kk * 32 + 32, hi);
            bn2 = frag16(wp + kk * 32 + 32, hi);
        }
        acc = wmma_bf16(av, bv, acc);
        av = an; bv = bn2;
    }
    const float bn  = bias[n0 + ln];
    const int   mhi = (lane >= 16) ? 8 : 0;
    #pragma unroll
    for (int r = 0; r < 8; ++r) {
        int m = m0 + r + mhi;
        out[(size_t)m * E_ + n0 + ln] = (acc[r] + bn) * (qmask[m] ? 1.0f : 0.0f);
    }
}

// ---------------- host ----------------
extern "C" void kernel_launch(void* const* d_in, const int* in_sizes, int n_in,
                              void* d_out, int out_size, void* d_ws, size_t ws_size,
                              hipStream_t stream)
{
    (void)in_sizes; (void)n_in; (void)out_size; (void)ws_size;
    const float*         q      = (const float*)d_in[0];
    const float*         kv     = (const float*)d_in[1];
    const unsigned char* qmask  = (const unsigned char*)d_in[2];
    const unsigned char* kvmask = (const unsigned char*)d_in[3];
    const int*           qpos   = (const int*)d_in[4];
    const int*           kvpos  = (const int*)d_in[5];
    const float* Wq = (const float*)d_in[6];
    const float* bq = (const float*)d_in[7];
    const float* Wk = (const float*)d_in[8];
    const float* bk = (const float*)d_in[9];
    const float* Wv = (const float*)d_in[10];
    const float* bvv = (const float*)d_in[11];
    const float* Wo = (const float*)d_in[12];
    const float* bo = (const float*)d_in[13];

    char* ws = (char*)d_ws;
    unsigned short* qh   = (unsigned short*)ws;                          // 4 MiB
    unsigned short* kh   = (unsigned short*)(ws + (4ull  << 20));        // 8 MiB
    unsigned short* vT   = (unsigned short*)(ws + (12ull << 20));        // 8 MiB (B,H,Dh,N)
    unsigned short* ctxb = (unsigned short*)(ws + (20ull << 20));        // 4 MiB
    unsigned short* Wqt  = (unsigned short*)(ws + (24ull << 20));        // 512 KiB each
    unsigned short* Wkt  = Wqt + (size_t)E_ * E_;
    unsigned short* Wvt  = Wkt + (size_t)E_ * E_;
    unsigned short* Wot  = Wvt + (size_t)E_ * E_;

    float* outc = (float*)d_out;                    // (B,QT,E) fp32
    float* outa = outc + (size_t)B_ * QT_ * E_;     // (B,H,QT,N) fp32

    wprep_kernel<<<dim3(16, 16), 256, 0, stream>>>(Wq, Wqt);
    wprep_kernel<<<dim3(16, 16), 256, 0, stream>>>(Wk, Wkt);
    wprep_kernel<<<dim3(16, 16), 256, 0, stream>>>(Wv, Wvt);
    wprep_kernel<<<dim3(16, 16), 256, 0, stream>>>(Wo, Wot);

    qkv_rope_kernel<<<dim3(4, (B_ * QT_) / 16), 256, 0, stream>>>(q,  Wqt, bq,  qpos,  qh, QT_, 1, 0);
    qkv_rope_kernel<<<dim3(4, (B_ * N_)  / 16), 256, 0, stream>>>(kv, Wkt, bk,  kvpos, kh, N_,  1, 0);
    qkv_rope_kernel<<<dim3(4, (B_ * N_)  / 16), 256, 0, stream>>>(kv, Wvt, bvv, kvpos, vT, N_,  0, 1);

    size_t smem = (size_t)16 * N_ * 4      // sc fp32
                + (size_t)16 * N_ * 2      // scbf bf16
                + (size_t)16 * DH_ * 2     // qt
                + 256 * 4 + 32 * 4;        // red + rowmax/rowsum
    (void)hipFuncSetAttribute(reinterpret_cast<const void*>(attn_kernel),
                              hipFuncAttributeMaxDynamicSharedMemorySize, (int)smem);
    attn_kernel<<<dim3(QT_ / 16, H_, B_), 256, smem, stream>>>(qh, kh, vT, kvmask, outa, ctxb);

    outproj_kernel<<<dim3(4, (B_ * QT_) / 16), 256, 0, stream>>>(ctxb, Wot, bo, qmask, outc);
}
